// ChebNet_59691455480077
// MI455X (gfx1250) — matrix operand order repaired and checked by reading, not compiled, not run on hardware
//
#include <hip/hip_runtime.h>

typedef float  v8f   __attribute__((ext_vector_type(8)));
typedef __bf16 v16bf __attribute__((ext_vector_type(16)));
typedef __bf16 v8bf  __attribute__((ext_vector_type(8)));
typedef __bf16 v4bf  __attribute__((ext_vector_type(4)));

#define NN  65536
#define NE  1048576
#define NG  64
#define NPG 1024

__device__ __forceinline__ void atomAddF(float* p, float v) {
  // native global_atomic_add_f32, relaxed, device scope (no CAS loop)
  __hip_atomic_fetch_add(p, v, __ATOMIC_RELAXED, __HIP_MEMORY_SCOPE_AGENT);
}

// ---------------- graph normalization ----------------
__global__ void k_zero4(float4* __restrict__ p, long n4) {
  long i = (long)blockIdx.x * blockDim.x + threadIdx.x;
  if (i < n4) p[i] = make_float4(0.f, 0.f, 0.f, 0.f);
}

__global__ void k_degree(const int* __restrict__ src, const int* __restrict__ dst,
                         float* __restrict__ deg) {
  int e = blockIdx.x * blockDim.x + threadIdx.x;
  if (e >= NE) return;
  int s = src[e];
  if (s != dst[e]) atomAddF(&deg[s], 1.f);
}

__global__ void k_dis(const float* __restrict__ deg, float* __restrict__ dis) {
  int i = blockIdx.x * blockDim.x + threadIdx.x;
  if (i >= NN) return;
  float d = deg[i];
  dis[i] = (d > 0.f) ? rsqrtf(fmaxf(d, 1.f)) : 0.f;
}

__global__ void k_edgew(const int* __restrict__ src, const int* __restrict__ dst,
                        const int* __restrict__ batch, const float* __restrict__ lam,
                        const float* __restrict__ dis, float* __restrict__ w) {
  int e = blockIdx.x * blockDim.x + threadIdx.x;
  if (e >= NE) return;
  int s = src[e], d = dst[e];
  float we = 0.f;
  if (s != d) we = -2.f * dis[s] * dis[d] / lam[batch[s]];
  w[e] = we;
}

__global__ void k_diag(const int* __restrict__ batch, const float* __restrict__ lam,
                       float* __restrict__ diag) {
  int i = blockIdx.x * blockDim.x + threadIdx.x;
  if (i >= NN) return;
  diag[i] = 2.f / lam[batch[i]] - 1.f;
}

// ---------------- bf16 staging ----------------
// f32 -> bf16 bulk convert (8 elements / thread, b128 store)
__global__ void k_cvt_bf16(const float* __restrict__ src, __bf16* __restrict__ dst, long n8) {
  long i = (long)blockIdx.x * blockDim.x + threadIdx.x;
  if (i >= n8) return;
  const float4* s = (const float4*)src + i * 2;
  float4 a = s[0], b = s[1];
  v8bf o;
  o[0] = (__bf16)a.x; o[1] = (__bf16)a.y; o[2] = (__bf16)a.z; o[3] = (__bf16)a.w;
  o[4] = (__bf16)b.x; o[5] = (__bf16)b.y; o[6] = (__bf16)b.z; o[7] = (__bf16)b.w;
  ((v8bf*)dst)[i] = o;
}

// W[5][Fin][Fout] f32  ->  WT[5][Fout][Fin] bf16  (one-time tiny transpose)
__global__ void k_cvt_wT(const float* __restrict__ W, __bf16* __restrict__ WT,
                         int Fin, int Fout, int total) {
  int i = blockIdx.x * blockDim.x + threadIdx.x;
  if (i >= total) return;
  int per = Fin * Fout;
  int t = i / per, r = i - t * per;
  int kk = r / Fout, n = r - kk * Fout;
  WT[(size_t)t * per + (size_t)n * Fin + kk] = (__bf16)W[i];
}

// ---------------- sparse Lhat pieces ----------------
__global__ void k_spmv(const float* __restrict__ v, const int* __restrict__ src,
                       const int* __restrict__ dst, const float* __restrict__ w,
                       float* __restrict__ agg, int chunks /* F/4 */, long total) {
  long idx = (long)blockIdx.x * blockDim.x + threadIdx.x;
  if (idx >= total) return;
  int e = (int)(idx / chunks);
  int c = (int)(idx - (long)e * chunks);
  float we = w[e];
  if (we == 0.f) return;
  int F = chunks << 2;
  const float4 vv = *(const float4*)(v + (long)src[e] * F + (c << 2));
  float* o = agg + (long)dst[e] * F + (c << 2);
  atomAddF(o + 0, we * vv.x);
  atomAddF(o + 1, we * vv.y);
  atomAddF(o + 2, we * vv.z);
  atomAddF(o + 3, we * vv.w);
}

// tnew = alpha*(agg + diag[row]*tcur) - (tprev ? tprev : 0); dual write f32 + bf16
__global__ void k_combine(float* __restrict__ tnew, __bf16* __restrict__ tnewbf,
                          const float* __restrict__ agg, const float* __restrict__ tcur,
                          const float* __restrict__ tprev, const float* __restrict__ diag,
                          float alpha, int F, long total4) {
  long i = (long)blockIdx.x * blockDim.x + threadIdx.x;
  if (i >= total4) return;
  long base = i << 2;
  int row = (int)(base / F);
  float dg = diag[row];
  float4 a = *(const float4*)(agg + base);
  float4 c = *(const float4*)(tcur + base);
  float4 r;
  r.x = alpha * (a.x + dg * c.x);
  r.y = alpha * (a.y + dg * c.y);
  r.z = alpha * (a.z + dg * c.z);
  r.w = alpha * (a.w + dg * c.w);
  if (tprev) {
    float4 p = *(const float4*)(tprev + base);
    r.x -= p.x; r.y -= p.y; r.z -= p.z; r.w -= p.w;
  }
  *(float4*)(tnew + base) = r;
  v4bf rb;
  rb[0] = (__bf16)r.x; rb[1] = (__bf16)r.y; rb[2] = (__bf16)r.z; rb[3] = (__bf16)r.w;
  *(v4bf*)(tnewbf + base) = rb;
}

__global__ void k_biasrelu(float* __restrict__ out, __bf16* __restrict__ outbf,
                           const float* __restrict__ b, int F, long total) {
  long i = (long)blockIdx.x * blockDim.x + threadIdx.x;
  if (i >= total) return;
  float v = fmaxf(out[i] + b[i % F], 0.f);
  out[i] = v;
  outbf[i] = (__bf16)v;
}

// ---------------- dense GEMM: C[M,N] (+)= A[M,K] @ W[K,N]  (A bf16, WT = W^T bf16) ----
// one wave computes a 16 x (NT*16) strip; K, NT compile-time -> fully unrolled WMMA chain
template <int K, int NT>
__global__ __launch_bounds__(128) void k_gemm(const __bf16* __restrict__ A,
                                              const __bf16* __restrict__ WT,
                                              float* __restrict__ C, int accumulate) {
  constexpr int N = NT * 16;
  int lane = threadIdx.x & 31;
  int wave = threadIdx.x >> 5;
  int half = lane >> 4;  // lanes 0-15 vs 16-31
  int l    = lane & 15;
  int m0   = (blockIdx.x * 4 + wave) << 4;
  int row  = m0 + l;

  v8f acc[NT];
  if (accumulate) {
#pragma unroll
    for (int nt = 0; nt < NT; ++nt)
#pragma unroll
      for (int j = 0; j < 8; ++j)
        acc[nt][j] = C[(size_t)(m0 + j + half * 8) * N + nt * 16 + l];
  } else {
#pragma unroll
    for (int nt = 0; nt < NT; ++nt)
#pragma unroll
      for (int j = 0; j < 8; ++j) acc[nt][j] = 0.f;
  }

  const __bf16* arow = A + (size_t)row * K;
#pragma unroll
  for (int k = 0; k < K; k += 32) {
    // A 16x32 bf16 fragment: two b128 loads (elements 0-7 and 8-15 per lane)
    v8bf alo = *(const v8bf*)(arow + k + half * 8);
    v8bf ahi = *(const v8bf*)(arow + k + 16 + half * 8);
    v16bf af = __builtin_shufflevector(alo, ahi, 0, 1, 2, 3, 4, 5, 6, 7,
                                       8, 9, 10, 11, 12, 13, 14, 15);
#pragma unroll
    for (int nt = 0; nt < NT; ++nt) {
      // B 32x16 fragment: column n = nt*16+l of W  ==  row of WT, contiguous in K
      const __bf16* bp = WT + (size_t)(nt * 16 + l) * K + k + half * 8;
      v8bf blo = *(const v8bf*)bp;
      v8bf bhi = *(const v8bf*)(bp + 16);
      v16bf bfrag = __builtin_shufflevector(blo, bhi, 0, 1, 2, 3, 4, 5, 6, 7,
                                            8, 9, 10, 11, 12, 13, 14, 15);
      acc[nt] = __builtin_amdgcn_wmma_f32_16x16x32_bf16(
          false, af, false, bfrag, (short)0, acc[nt], false, false);
    }
  }

#pragma unroll
  for (int nt = 0; nt < NT; ++nt)
#pragma unroll
    for (int j = 0; j < 8; ++j)
      C[(size_t)(m0 + j + half * 8) * N + nt * 16 + l] = acc[nt][j];
}

// ---------------- pooling + FC head ----------------
__global__ void k_pool(const float* __restrict__ h, float* __restrict__ g) {
  int graph = blockIdx.x;  // 64
  int f = threadIdx.x;     // 64 features
  const float* p = h + (long)graph * NPG * 64 + f;
  float s = 0.f;
  for (int i = 0; i < NPG; ++i) s += p[(long)i * 64];
  g[graph * 64 + f] = s * (1.f / (float)NPG);
}

__global__ void k_fc(const float* __restrict__ g, const float* __restrict__ w1,
                     const float* __restrict__ b1, const float* __restrict__ w2,
                     const float* __restrict__ b2, float* __restrict__ out) {
  __shared__ float t[NG * 10];
  int i = threadIdx.x / 10;
  int j = threadIdx.x % 10;
  float s = b1[j];
  for (int k = 0; k < 64; ++k) s += g[i * 64 + k] * w1[k * 10 + j];
  t[i * 10 + j] = fmaxf(s, 0.f);
  __syncthreads();
  float o = b2[j];
  for (int k = 0; k < 10; ++k) o += t[i * 10 + k] * w2[k * 10 + j];
  out[i * 10 + j] = o;
}

// ---------------- host-side orchestration ----------------
static inline long cdiv(long a, long b) { return (a + b - 1) / b; }

static void zero_buf(float* p, long n, hipStream_t s) {
  long n4 = n >> 2;
  k_zero4<<<cdiv(n4, 256), 256, 0, s>>>((float4*)p, n4);
}

static void gemm(const __bf16* A, const __bf16* WT, float* C, int K, int N, int acc,
                 hipStream_t s) {
  dim3 grid(NN / 64);  // 4 waves/block, 16 rows/wave
  if (K == 128 && N == 32)      k_gemm<128, 2><<<grid, 128, 0, s>>>(A, WT, C, acc);
  else if (K == 32 && N == 64)  k_gemm<32, 4><<<grid, 128, 0, s>>>(A, WT, C, acc);
  else if (K == 64 && N == 64)  k_gemm<64, 4><<<grid, 128, 0, s>>>(A, WT, C, acc);
}

static void cheb_layer(const float* X, const __bf16* Xbf, int Fin,
                       const __bf16* WT, const float* bias, int Fout,
                       float* out, __bf16* outbf,
                       float* tA, __bf16* tAbf, float* tB, __bf16* tBbf,
                       float* tC, __bf16* tCbf, float* agg,
                       const int* src, const int* dst, const float* ew, const float* diag,
                       hipStream_t s) {
  long nF  = (long)NN * Fin;
  long spT = (long)NE * (Fin >> 2);
  long c4  = nF >> 2;
  size_t wstep = (size_t)Fin * Fout;  // per Chebyshev term, in elements

  // k = 0
  gemm(Xbf, WT, out, Fin, Fout, 0, s);
  // Tx1 = Lhat(X)
  zero_buf(agg, nF, s);
  k_spmv<<<cdiv(spT, 256), 256, 0, s>>>(X, src, dst, ew, agg, Fin >> 2, spT);
  k_combine<<<cdiv(c4, 256), 256, 0, s>>>(tA, tAbf, agg, X, nullptr, diag, 1.f, Fin, c4);
  gemm(tAbf, WT + wstep, out, Fin, Fout, 1, s);

  const float*  prev   = X;
  const float*  cur    = tA;
  const __bf16* curbf  = tAbf;
  float*  nxt[3]   = { tB, tC, tA };
  __bf16* nxtbf[3] = { tBbf, tCbf, tAbf };
  for (int k = 2; k < 5; ++k) {
    float*  t2   = nxt[k - 2];
    __bf16* t2bf = nxtbf[k - 2];
    zero_buf(agg, nF, s);
    k_spmv<<<cdiv(spT, 256), 256, 0, s>>>(cur, src, dst, ew, agg, Fin >> 2, spT);
    k_combine<<<cdiv(c4, 256), 256, 0, s>>>(t2, t2bf, agg, cur, prev, diag, 2.f, Fin, c4);
    gemm(t2bf, WT + (size_t)k * wstep, out, Fin, Fout, 1, s);
    prev = cur;
    cur = t2;
    curbf = t2bf;
  }
  (void)curbf;
  long oT = (long)NN * Fout;
  k_biasrelu<<<cdiv(oT, 256), 256, 0, s>>>(out, outbf, bias, Fout, oT);
}

extern "C" void kernel_launch(void* const* d_in, const int* in_sizes, int n_in,
                              void* d_out, int out_size, void* d_ws, size_t ws_size,
                              hipStream_t stream) {
  const float* x    = (const float*)d_in[0];
  const int*   edge = (const int*)d_in[1];
  const int*   batch = (const int*)d_in[2];
  const float* lam  = (const float*)d_in[3];
  const float* W1   = (const float*)d_in[4];
  const float* b1   = (const float*)d_in[5];
  const float* W2   = (const float*)d_in[6];
  const float* b2   = (const float*)d_in[7];
  const float* W3   = (const float*)d_in[8];
  const float* b3   = (const float*)d_in[9];
  const float* fcW1 = (const float*)d_in[10];
  const float* fcb1 = (const float*)d_in[11];
  const float* fcW2 = (const float*)d_in[12];
  const float* fcb2 = (const float*)d_in[13];
  float* out = (float*)d_out;

  const int* src = edge;
  const int* dst = edge + NE;

  // workspace carve-up (float units; bf16 buffers take n/2 float slots)
  float* ws = (float*)d_ws;
  size_t off = 0;
  auto allocf = [&](size_t n) { float* p = ws + off; off += n; return p; };
  auto allocb = [&](size_t n) { __bf16* p = (__bf16*)(ws + off); off += (n + 1) / 2; return p; };

  float*  deg  = allocf(NN);
  float*  dis  = allocf(NN);
  float*  diag = allocf(NN);
  float*  ew   = allocf(NE);
  float*  tA   = allocf((size_t)NN * 128);
  float*  tB   = allocf((size_t)NN * 128);
  float*  tC   = allocf((size_t)NN * 128);
  float*  agg  = allocf((size_t)NN * 128);
  float*  h1   = allocf((size_t)NN * 32);
  float*  h2   = allocf((size_t)NN * 64);
  float*  h3   = allocf((size_t)NN * 64);
  float*  gp   = allocf((size_t)NG * 64);
  __bf16* xbf  = allocb((size_t)NN * 128);
  __bf16* tAbf = allocb((size_t)NN * 128);
  __bf16* tBbf = allocb((size_t)NN * 128);
  __bf16* tCbf = allocb((size_t)NN * 128);
  __bf16* h1bf = allocb((size_t)NN * 32);
  __bf16* h2bf = allocb((size_t)NN * 64);
  __bf16* h3bf = allocb((size_t)NN * 64);
  __bf16* WT1  = allocb((size_t)5 * 128 * 32);
  __bf16* WT2  = allocb((size_t)5 * 32 * 64);
  __bf16* WT3  = allocb((size_t)5 * 64 * 64);
  (void)ws_size; (void)n_in; (void)in_sizes; (void)out_size;

  // ---- edge normalization ----
  zero_buf(deg, NN, stream);
  k_degree<<<cdiv(NE, 256), 256, 0, stream>>>(src, dst, deg);
  k_dis<<<cdiv(NN, 256), 256, 0, stream>>>(deg, dis);
  k_edgew<<<cdiv(NE, 256), 256, 0, stream>>>(src, dst, batch, lam, dis, ew);
  k_diag<<<cdiv(NN, 256), 256, 0, stream>>>(batch, lam, diag);

  // ---- bf16 staging of inputs / weights ----
  k_cvt_bf16<<<cdiv((long)NN * 128 / 8, 256), 256, 0, stream>>>(x, xbf, (long)NN * 128 / 8);
  k_cvt_wT<<<cdiv(5 * 128 * 32, 256), 256, 0, stream>>>(W1, WT1, 128, 32, 5 * 128 * 32);
  k_cvt_wT<<<cdiv(5 * 32 * 64, 256), 256, 0, stream>>>(W2, WT2, 32, 64, 5 * 32 * 64);
  k_cvt_wT<<<cdiv(5 * 64 * 64, 256), 256, 0, stream>>>(W3, WT3, 64, 64, 5 * 64 * 64);

  // ---- three ChebConv layers ----
  cheb_layer(x,  xbf,  128, WT1, b1, 32, h1, h1bf, tA, tAbf, tB, tBbf, tC, tCbf, agg,
             src, dst, ew, diag, stream);
  cheb_layer(h1, h1bf,  32, WT2, b2, 64, h2, h2bf, tA, tAbf, tB, tBbf, tC, tCbf, agg,
             src, dst, ew, diag, stream);
  cheb_layer(h2, h2bf,  64, WT3, b3, 64, h3, h3bf, tA, tAbf, tB, tBbf, tC, tCbf, agg,
             src, dst, ew, diag, stream);

  // ---- pool + FC head ----
  k_pool<<<NG, 64, 0, stream>>>(h3, gp);
  k_fc<<<1, NG * 10, 0, stream>>>(gp, fcW1, fcb1, fcW2, fcb2, out);
}
